// Transformer_48765058679014
// MI455X (gfx1250) — compile-verified, matrix operand order checked by reading, not run on hardware
//
#include <hip/hip_runtime.h>
#include <hip/hip_bf16.h>

// ---------------- problem constants ----------------
#define BB     4
#define SS     512
#define TT     512
#define VV     32000
#define HH     512
#define FF     2048
#define NLAYER 6
#define NHEAD  8
#define DH     64          // HH / NHEAD
#define PADTOK 0
#define BS     (BB * SS)   // 2048 rows

typedef unsigned short u16;
typedef __attribute__((ext_vector_type(16))) __bf16 v16bf;
typedef __attribute__((ext_vector_type(8)))  float  v8f;

__device__ __forceinline__ u16 f2bf(float f) {
    unsigned u = __float_as_uint(f);
    u += 0x7FFFu + ((u >> 16) & 1u);       // round-to-nearest-even
    return (u16)(u >> 16);
}

// CDNA5 async global->LDS copy (16B per lane), tracked by ASYNCcnt
__device__ __forceinline__ void async_ld16(unsigned ldsoff, const void* gptr) {
    asm volatile("global_load_async_to_lds_b128 %0, %1, off"
                 :: "v"(ldsoff), "v"((unsigned long long)(uintptr_t)gptr)
                 : "memory");
}
__device__ __forceinline__ void wait_async0() {
    asm volatile("s_wait_asynccnt 0x0" ::: "memory");
}

// ---------------- f32 -> bf16 conversion (no transpose) ----------------
__global__ __launch_bounds__(256) void k_cvt(const float* __restrict__ src,
                                             u16* __restrict__ dst, int n) {
    for (int i = blockIdx.x * 256 + threadIdx.x; i < n; i += gridDim.x * 256)
        dst[i] = f2bf(src[i]);
}

// ---------------- f32 [L][K][N] -> bf16 [L][N][K] (weight transpose) ----------------
__global__ __launch_bounds__(256) void k_cvt_t(const float* __restrict__ src,
                                               u16* __restrict__ dst, int Kd, int Nd) {
    __shared__ float tile[32][33];
    const int l = blockIdx.z;
    src += (size_t)l * Kd * Nd;
    dst += (size_t)l * Kd * Nd;
    const int kt = blockIdx.y * 32, nt = blockIdx.x * 32;
    for (int i = threadIdx.y; i < 32; i += 8)
        tile[i][threadIdx.x] = src[(size_t)(kt + i) * Nd + nt + threadIdx.x];
    __syncthreads();
    for (int i = threadIdx.y; i < 32; i += 8)
        dst[(size_t)(nt + i) * Kd + kt + threadIdx.x] = f2bf(tile[threadIdx.x][i]);
}

// ---------------- bf16 [B*S][H] -> [B][H][S] (V transpose for AV gemm) ----------------
__global__ __launch_bounds__(256) void k_trans_v(const u16* __restrict__ v,
                                                 u16* __restrict__ vt) {
    __shared__ u16 tile[32][33];
    const int b = blockIdx.z;
    const int st = blockIdx.x * 32, ct = blockIdx.y * 32;
    for (int i = threadIdx.y; i < 32; i += 8)
        tile[i][threadIdx.x] = v[((size_t)b * SS + st + i) * HH + ct + threadIdx.x];
    __syncthreads();
    for (int i = threadIdx.y; i < 32; i += 8)
        vt[(size_t)b * HH * SS + (size_t)(ct + i) * SS + st + threadIdx.x] = tile[threadIdx.x][i];
}

// ---------------- embedding + sinusoidal PE ----------------
__global__ __launch_bounds__(256) void k_embed(const int* __restrict__ tok,
                                               const float* __restrict__ emb,
                                               float* __restrict__ x, int L, int shift) {
    const int bs = blockIdx.x;
    const int b = bs / L, s = bs % L;
    int t = 0; bool zero;
    if (shift) {
        if (s == 0) { zero = true; }
        else { t = tok[b * L + s - 1]; zero = (t == PADTOK); }
    } else {
        t = tok[b * L + s]; zero = (t == PADTOK);
    }
    const float scale = 22.62741699796952f;             // sqrt(512)
    const float li = 9.210340371976184f / (HH / 2 - 1); // ln(10000)/(nt-1)
    for (int j = threadIdx.x; j < HH; j += 256) {
        float pe;
        if (j < HH / 2) pe = __sinf((float)s * __expf(-li * (float)j));
        else            pe = __cosf((float)s * __expf(-li * (float)(j - HH / 2)));
        float e = zero ? 0.f : emb[(size_t)t * HH + j];
        x[(size_t)bs * HH + j] = e * scale + pe;
    }
}

// ---------------- LayerNorm (f32 in, bf16 out) ----------------
__global__ __launch_bounds__(256) void k_ln(const float* __restrict__ x,
                                            const float* __restrict__ g,
                                            const float* __restrict__ bta,
                                            u16* __restrict__ y) {
    __shared__ float red[256];
    const int row = blockIdx.x;
    const float* xr = x + (size_t)row * HH;
    float s = 0.f;
    for (int j = threadIdx.x; j < HH; j += 256) s += xr[j];
    red[threadIdx.x] = s; __syncthreads();
    for (int o = 128; o > 0; o >>= 1) { if (threadIdx.x < o) red[threadIdx.x] += red[threadIdx.x + o]; __syncthreads(); }
    const float mean = red[0] * (1.f / HH);
    __syncthreads();
    float vs = 0.f;
    for (int j = threadIdx.x; j < HH; j += 256) { float d = xr[j] - mean; vs += d * d; }
    red[threadIdx.x] = vs; __syncthreads();
    for (int o = 128; o > 0; o >>= 1) { if (threadIdx.x < o) red[threadIdx.x] += red[threadIdx.x + o]; __syncthreads(); }
    const float rstd = rsqrtf(red[0] * (1.f / HH) + 1e-6f);
    for (int j = threadIdx.x; j < HH; j += 256)
        y[(size_t)row * HH + j] = f2bf((xr[j] - mean) * rstd * g[j] + bta[j]);
}

// ---------------- masked softmax over f32 scores -> bf16 probs ----------------
__global__ __launch_bounds__(256) void k_softmax(const float* __restrict__ scores,
                                                 u16* __restrict__ probs,
                                                 const int* __restrict__ tok,
                                                 int Tq, int Tk, int causal, int nh) {
    __shared__ float red[256];
    const int row = blockIdx.x;
    const int q = row % Tq;
    const int b = row / (Tq * nh);
    const float* sr = scores + (size_t)row * Tk;
    float v[4];
    float mx = -3.0e38f;
    int c = 0;
    for (int j = threadIdx.x; j < Tk; j += 256) {
        float sv = sr[j];
        bool m = causal ? (j > q) : (tok[b * Tk + j] == PADTOK);
        sv = m ? -1.0e9f : sv;
        v[c++] = sv;
        mx = fmaxf(mx, sv);
    }
    red[threadIdx.x] = mx; __syncthreads();
    for (int o = 128; o > 0; o >>= 1) { if (threadIdx.x < o) red[threadIdx.x] = fmaxf(red[threadIdx.x], red[threadIdx.x + o]); __syncthreads(); }
    mx = red[0]; __syncthreads();
    float sum = 0.f; c = 0;
    for (int j = threadIdx.x; j < Tk; j += 256) { float e = __expf(v[c] - mx); v[c++] = e; sum += e; }
    red[threadIdx.x] = sum; __syncthreads();
    for (int o = 128; o > 0; o >>= 1) { if (threadIdx.x < o) red[threadIdx.x] += red[threadIdx.x + o]; __syncthreads(); }
    const float inv = 1.f / red[0];
    c = 0;
    for (int j = threadIdx.x; j < Tk; j += 256) probs[(size_t)row * Tk + j] = f2bf(v[c++] * inv);
}

// ---------------- WMMA bf16 GEMM (B given as [N][K] row-major) ----------------
// Macro tile 128x64; 8 waves in 4x2; each wave owns a 32x32 patch = 4 WMMA accums.
// Double-buffered LDS filled by GLOBAL_LOAD_ASYNC_TO_LDS_B128 (ASYNCcnt).
__global__ __launch_bounds__(256) void k_gemm_bf16(
    int M, int N, int K,
    const u16* __restrict__ A, int lda, long sAb, long sAh,
    const u16* __restrict__ B, int ldb, long sBb, long sBh,
    const float* __restrict__ bias, int relu,
    const float* __restrict__ resid, float alpha,
    float* __restrict__ Cf, int ldc,
    u16* __restrict__ Cb, int ldcb,
    long sCzb, long sCzh, int nh)
{
    __shared__ __align__(16) u16 lA[2][128 * 32];
    __shared__ __align__(16) u16 lB[2][64 * 32];

    const int z = blockIdx.z;
    const int zb = z / nh, zh = z % nh;
    A += (size_t)zb * sAb + (size_t)zh * sAh;
    B += (size_t)zb * sBb + (size_t)zh * sBh;
    const long coff = (long)zb * sCzb + (long)zh * sCzh;

    const int tid = threadIdx.x;
    const int lane = tid & 31;
    const int wave = tid >> 5;
    const int wm = wave & 3;          // 0..3 -> 32-row group
    const int wn = wave >> 2;         // 0..1 -> 32-col group
    const int bm = blockIdx.y * 128;
    const int bn = blockIdx.x * 64;

    v8f acc[2][2];
    acc[0][0] = v8f{0.f,0.f,0.f,0.f,0.f,0.f,0.f,0.f};
    acc[0][1] = acc[0][0]; acc[1][0] = acc[0][0]; acc[1][1] = acc[0][0];

    // staging: each thread moves 16B chunks; A tile is 128x32 (2 chunks), B 64x32 (1)
    const int r8 = tid >> 2, c8 = (tid & 3) << 3;
    const u16* gA0 = A + (size_t)(bm + r8) * lda + c8;
    const u16* gA1 = A + (size_t)(bm + 64 + r8) * lda + c8;
    const u16* gB  = B + (size_t)(bn + r8) * ldb + c8;
    const unsigned offA0 = (unsigned)(uintptr_t)&lA[0][r8 * 32 + c8];
    const unsigned offA1 = (unsigned)(uintptr_t)&lA[0][(64 + r8) * 32 + c8];
    const unsigned offB0 = (unsigned)(uintptr_t)&lB[0][r8 * 32 + c8];
    const unsigned bufA = 128 * 32 * 2;   // bytes per A buffer
    const unsigned bufB = 64 * 32 * 2;    // bytes per B buffer

    const int mrow = lane & 15;
    const int ncol = lane & 15;
    const int kb = (lane >> 4) << 3;      // K base 0 or 8 per lane half

    async_ld16(offA0, gA0);
    async_ld16(offA1, gA1);
    async_ld16(offB0, gB);

    for (int k0 = 0; k0 < K; k0 += 32) {
        const int buf = (k0 >> 5) & 1;
        wait_async0();
        __syncthreads();
        if (k0 + 32 < K) {
            async_ld16(offA0 + (buf ^ 1) * bufA, gA0 + k0 + 32);
            async_ld16(offA1 + (buf ^ 1) * bufA, gA1 + k0 + 32);
            async_ld16(offB0 + (buf ^ 1) * bufB, gB + k0 + 32);
        }
        // fragments: lane's 16 elements = K[kb..kb+7] ++ K[16+kb..16+kb+7]
        union V16 { uint4 q[2]; v16bf b; } a0, a1, b0, b1;
        const u16* pa = &lA[buf][(wm * 32 + mrow) * 32];
        const u16* pb = &lB[buf][(wn * 32 + ncol) * 32];
        a0.q[0] = *(const uint4*)&pa[kb];            a0.q[1] = *(const uint4*)&pa[16 + kb];
        a1.q[0] = *(const uint4*)&pa[16 * 32 + kb];  a1.q[1] = *(const uint4*)&pa[16 * 32 + 16 + kb];
        b0.q[0] = *(const uint4*)&pb[kb];            b0.q[1] = *(const uint4*)&pb[16 + kb];
        b1.q[0] = *(const uint4*)&pb[16 * 32 + kb];  b1.q[1] = *(const uint4*)&pb[16 * 32 + 16 + kb];
        acc[0][0] = __builtin_amdgcn_wmma_f32_16x16x32_bf16(false, a0.b, false, b0.b,
                                                            (short)0, acc[0][0], false, false);
        acc[0][1] = __builtin_amdgcn_wmma_f32_16x16x32_bf16(false, a0.b, false, b1.b,
                                                            (short)0, acc[0][1], false, false);
        acc[1][0] = __builtin_amdgcn_wmma_f32_16x16x32_bf16(false, a1.b, false, b0.b,
                                                            (short)0, acc[1][0], false, false);
        acc[1][1] = __builtin_amdgcn_wmma_f32_16x16x32_bf16(false, a1.b, false, b1.b,
                                                            (short)0, acc[1][1], false, false);
    }

    // epilogue: VGPR r holds M = r (lanes 0-15) / 8+r (lanes 16-31), N = lane&15
    const int rbase = (lane >> 4) << 3;
#pragma unroll
    for (int mi = 0; mi < 2; ++mi) {
#pragma unroll
        for (int ni = 0; ni < 2; ++ni) {
#pragma unroll
            for (int r = 0; r < 8; ++r) {
                const int row = bm + wm * 32 + mi * 16 + rbase + r;
                const int col = bn + wn * 32 + ni * 16 + ncol;
                float v = acc[mi][ni][r] * alpha;
                if (bias)  v += bias[col];
                if (relu)  v = v > 0.f ? v : 0.f;
                if (resid) v += resid[coff + (size_t)row * ldc + col];
                if (Cf) Cf[coff + (size_t)row * ldc + col] = v;
                if (Cb) Cb[coff + (size_t)row * ldcb + col] = f2bf(v);
            }
        }
    }
}

// ======================= host orchestration =======================
extern "C" void kernel_launch(void* const* d_in, const int* in_sizes, int n_in,
                              void* d_out, int out_size, void* d_ws, size_t ws_size,
                              hipStream_t stream) {
    (void)in_sizes; (void)n_in; (void)out_size; (void)ws_size;

    const int*   inputs  = (const int*)d_in[0];
    const int*   targets = (const int*)d_in[1];
    const float* emb     = (const float*)d_in[2];
    const float* enc_ln1_g = (const float*)d_in[3];
    const float* enc_ln1_b = (const float*)d_in[4];
    const float* enc_wq = (const float*)d_in[5];
    const float* enc_wk = (const float*)d_in[6];
    const float* enc_wv = (const float*)d_in[7];
    const float* enc_wo = (const float*)d_in[8];
    const float* enc_ln2_g = (const float*)d_in[9];
    const float* enc_ln2_b = (const float*)d_in[10];
    const float* enc_w1 = (const float*)d_in[11];
    const float* enc_b1 = (const float*)d_in[12];
    const float* enc_w2 = (const float*)d_in[13];
    const float* enc_b2 = (const float*)d_in[14];
    const float* enc_lnf_g = (const float*)d_in[15];
    const float* enc_lnf_b = (const float*)d_in[16];
    const float* dec_ln1_g = (const float*)d_in[17];
    const float* dec_ln1_b = (const float*)d_in[18];
    const float* dec_wq1 = (const float*)d_in[19];
    const float* dec_wk1 = (const float*)d_in[20];
    const float* dec_wv1 = (const float*)d_in[21];
    const float* dec_wo1 = (const float*)d_in[22];
    const float* dec_ln2_g = (const float*)d_in[23];
    const float* dec_ln2_b = (const float*)d_in[24];
    const float* dec_wq2 = (const float*)d_in[25];
    const float* dec_wk2 = (const float*)d_in[26];
    const float* dec_wv2 = (const float*)d_in[27];
    const float* dec_wo2 = (const float*)d_in[28];
    const float* dec_ln3_g = (const float*)d_in[29];
    const float* dec_ln3_b = (const float*)d_in[30];
    const float* dec_w1 = (const float*)d_in[31];
    const float* dec_b1 = (const float*)d_in[32];
    const float* dec_w2 = (const float*)d_in[33];
    const float* dec_b2 = (const float*)d_in[34];
    const float* dec_lnf_g = (const float*)d_in[35];
    const float* dec_lnf_b = (const float*)d_in[36];

    // ---- workspace carve ----
    char* ws = (char*)d_ws;
    size_t off = 0;
    auto alloc = [&](size_t bytes) -> void* {
        void* p = ws + off;
        off += (bytes + 255) & ~(size_t)255;
        return p;
    };
    u16* emb_bf   = (u16*)alloc((size_t)VV * HH * 2);          // [V][H] == [N][K]
    u16* ewqt     = (u16*)alloc((size_t)NLAYER * HH * HH * 2); // all weights stored [N][K]
    u16* ewkt     = (u16*)alloc((size_t)NLAYER * HH * HH * 2);
    u16* ewvt     = (u16*)alloc((size_t)NLAYER * HH * HH * 2);
    u16* ewot     = (u16*)alloc((size_t)NLAYER * HH * HH * 2);
    u16* ew1t     = (u16*)alloc((size_t)NLAYER * HH * FF * 2);
    u16* ew2t     = (u16*)alloc((size_t)NLAYER * FF * HH * 2);
    u16* dwq1t    = (u16*)alloc((size_t)NLAYER * HH * HH * 2);
    u16* dwk1t    = (u16*)alloc((size_t)NLAYER * HH * HH * 2);
    u16* dwv1t    = (u16*)alloc((size_t)NLAYER * HH * HH * 2);
    u16* dwo1t    = (u16*)alloc((size_t)NLAYER * HH * HH * 2);
    u16* dwq2t    = (u16*)alloc((size_t)NLAYER * HH * HH * 2);
    u16* dwk2t    = (u16*)alloc((size_t)NLAYER * HH * HH * 2);
    u16* dwv2t    = (u16*)alloc((size_t)NLAYER * HH * HH * 2);
    u16* dwo2t    = (u16*)alloc((size_t)NLAYER * HH * HH * 2);
    u16* dw1t     = (u16*)alloc((size_t)NLAYER * HH * FF * 2);
    u16* dw2t     = (u16*)alloc((size_t)NLAYER * FF * HH * 2);
    float* x_enc  = (float*)alloc((size_t)BS * HH * 4);
    float* x_dec  = (float*)alloc((size_t)BS * HH * 4);
    u16* y_bf     = (u16*)alloc((size_t)BS * HH * 2);
    u16* q_bf     = (u16*)alloc((size_t)BS * HH * 2);
    u16* k_bf     = (u16*)alloc((size_t)BS * HH * 2);
    u16* v_bf     = (u16*)alloc((size_t)BS * HH * 2);
    u16* vt_bf    = (u16*)alloc((size_t)BS * HH * 2);          // [B][H][S]
    u16* o_bf     = (u16*)alloc((size_t)BS * HH * 2);
    u16* encout_bf= (u16*)alloc((size_t)BS * HH * 2);
    u16* h1_bf    = (u16*)alloc((size_t)BS * FF * 2);
    float* scores = (float*)alloc((size_t)BB * NHEAD * SS * SS * 4);
    u16* probs    = (u16*)alloc((size_t)BB * NHEAD * SS * SS * 2);

    auto cvt = [&](const float* s, u16* d, long n) {
        int blocks = (int)((n + 256 * 8 - 1) / (256 * 8));
        if (blocks > 4096) blocks = 4096;
        k_cvt<<<dim3(blocks), dim3(256), 0, stream>>>(s, d, (int)n);
    };
    auto cvtT = [&](const float* s, u16* d, int Kd, int Nd) {
        k_cvt_t<<<dim3(Nd / 32, Kd / 32, NLAYER), dim3(32, 8), 0, stream>>>(s, d, Kd, Nd);
    };
    auto gemm = [&](int M, int N, int K,
                    const u16* A, int lda, long sAb, long sAh,
                    const u16* B, int ldb, long sBb, long sBh,
                    const float* bias, int relu, const float* resid, float alpha,
                    float* Cf, int ldc, u16* Cb, int ldcb,
                    long sCzb, long sCzh, int Z, int nh) {
        dim3 grid(N / 64, M / 128, Z);
        k_gemm_bf16<<<grid, dim3(256), 0, stream>>>(M, N, K, A, lda, sAb, sAh,
                                                    B, ldb, sBb, sBh,
                                                    bias, relu, resid, alpha,
                                                    Cf, ldc, Cb, ldcb, sCzb, sCzh, nh);
    };
    auto transV = [&](const u16* v, u16* vt) {
        k_trans_v<<<dim3(SS / 32, HH / 32, BB), dim3(32, 8), 0, stream>>>(v, vt);
    };

    const long HH2 = (long)HH * HH, HF = (long)HH * FF;
    const long SH = (long)SS * HH, SS2 = (long)SS * SS;
    const long HS = (long)HH * SS, DS = (long)DH * SS;
    const float qscale = 0.125f; // (H/NH)^-0.5

    // ---- weight conversion (all GEMM B-operands become [N][K]) ----
    cvt(emb, emb_bf, (long)VV * HH);
    cvtT(enc_wq, ewqt, HH, HH);  cvtT(enc_wk, ewkt, HH, HH);
    cvtT(enc_wv, ewvt, HH, HH);  cvtT(enc_wo, ewot, HH, HH);
    cvtT(enc_w1, ew1t, HH, FF);  cvtT(enc_w2, ew2t, FF, HH);
    cvtT(dec_wq1, dwq1t, HH, HH); cvtT(dec_wk1, dwk1t, HH, HH);
    cvtT(dec_wv1, dwv1t, HH, HH); cvtT(dec_wo1, dwo1t, HH, HH);
    cvtT(dec_wq2, dwq2t, HH, HH); cvtT(dec_wk2, dwk2t, HH, HH);
    cvtT(dec_wv2, dwv2t, HH, HH); cvtT(dec_wo2, dwo2t, HH, HH);
    cvtT(dec_w1, dw1t, HH, FF);   cvtT(dec_w2, dw2t, FF, HH);

    // ================= encoder =================
    k_embed<<<dim3(BS), dim3(256), 0, stream>>>(inputs, emb, x_enc, SS, 0);
    for (int i = 0; i < NLAYER; ++i) {
        k_ln<<<dim3(BS), dim3(256), 0, stream>>>(x_enc, enc_ln1_g + i * HH, enc_ln1_b + i * HH, y_bf);
        gemm(BS, HH, HH, y_bf, HH, 0, 0, ewqt + (size_t)i * HH2, HH, 0, 0,
             nullptr, 0, nullptr, qscale, nullptr, 0, q_bf, HH, 0, 0, 1, 1);
        gemm(BS, HH, HH, y_bf, HH, 0, 0, ewkt + (size_t)i * HH2, HH, 0, 0,
             nullptr, 0, nullptr, 1.f, nullptr, 0, k_bf, HH, 0, 0, 1, 1);
        gemm(BS, HH, HH, y_bf, HH, 0, 0, ewvt + (size_t)i * HH2, HH, 0, 0,
             nullptr, 0, nullptr, 1.f, nullptr, 0, v_bf, HH, 0, 0, 1, 1);
        transV(v_bf, vt_bf);
        // scores = q @ k^T  (K matrix rows are already [N][K])
        gemm(SS, SS, DH, q_bf, HH, SH, DH, k_bf, HH, SH, DH,
             nullptr, 0, nullptr, 1.f, scores, SS, nullptr, 0,
             (long)NHEAD * SS2, SS2, BB * NHEAD, NHEAD);
        k_softmax<<<dim3(BB * NHEAD * SS), dim3(256), 0, stream>>>(scores, probs, inputs, SS, SS, 0, NHEAD);
        // o = probs @ v  (B = V^T stored [B][H][S])
        gemm(SS, DH, SS, probs, SS, (long)NHEAD * SS2, SS2, vt_bf, SS, HS, DS,
             nullptr, 0, nullptr, 1.f, nullptr, 0, o_bf, HH, SH, DH, BB * NHEAD, NHEAD);
        gemm(BS, HH, HH, o_bf, HH, 0, 0, ewot + (size_t)i * HH2, HH, 0, 0,
             nullptr, 0, x_enc, 1.f, x_enc, HH, nullptr, 0, 0, 0, 1, 1);
        k_ln<<<dim3(BS), dim3(256), 0, stream>>>(x_enc, enc_ln2_g + i * HH, enc_ln2_b + i * HH, y_bf);
        gemm(BS, FF, HH, y_bf, HH, 0, 0, ew1t + (size_t)i * HF, HH, 0, 0,
             enc_b1 + (size_t)i * FF, 1, nullptr, 1.f, nullptr, 0, h1_bf, FF, 0, 0, 1, 1);
        gemm(BS, HH, FF, h1_bf, FF, 0, 0, ew2t + (size_t)i * HF, FF, 0, 0,
             enc_b2 + (size_t)i * HH, 0, x_enc, 1.f, x_enc, HH, nullptr, 0, 0, 0, 1, 1);
    }
    k_ln<<<dim3(BS), dim3(256), 0, stream>>>(x_enc, enc_lnf_g, enc_lnf_b, encout_bf);

    // ================= decoder =================
    k_embed<<<dim3(BS), dim3(256), 0, stream>>>(targets, emb, x_dec, TT, 1);
    for (int i = 0; i < NLAYER; ++i) {
        // --- masked self-attention ---
        k_ln<<<dim3(BS), dim3(256), 0, stream>>>(x_dec, dec_ln1_g + i * HH, dec_ln1_b + i * HH, y_bf);
        gemm(BS, HH, HH, y_bf, HH, 0, 0, dwq1t + (size_t)i * HH2, HH, 0, 0,
             nullptr, 0, nullptr, qscale, nullptr, 0, q_bf, HH, 0, 0, 1, 1);
        gemm(BS, HH, HH, y_bf, HH, 0, 0, dwk1t + (size_t)i * HH2, HH, 0, 0,
             nullptr, 0, nullptr, 1.f, nullptr, 0, k_bf, HH, 0, 0, 1, 1);
        gemm(BS, HH, HH, y_bf, HH, 0, 0, dwv1t + (size_t)i * HH2, HH, 0, 0,
             nullptr, 0, nullptr, 1.f, nullptr, 0, v_bf, HH, 0, 0, 1, 1);
        transV(v_bf, vt_bf);
        gemm(TT, TT, DH, q_bf, HH, SH, DH, k_bf, HH, SH, DH,
             nullptr, 0, nullptr, 1.f, scores, TT, nullptr, 0,
             (long)NHEAD * SS2, SS2, BB * NHEAD, NHEAD);
        k_softmax<<<dim3(BB * NHEAD * TT), dim3(256), 0, stream>>>(scores, probs, targets, TT, TT, 1, NHEAD);
        gemm(TT, DH, TT, probs, TT, (long)NHEAD * SS2, SS2, vt_bf, SS, HS, DS,
             nullptr, 0, nullptr, 1.f, nullptr, 0, o_bf, HH, SH, DH, BB * NHEAD, NHEAD);
        gemm(BS, HH, HH, o_bf, HH, 0, 0, dwo1t + (size_t)i * HH2, HH, 0, 0,
             nullptr, 0, x_dec, 1.f, x_dec, HH, nullptr, 0, 0, 0, 1, 1);
        // --- cross-attention (KV from encoder output) ---
        k_ln<<<dim3(BS), dim3(256), 0, stream>>>(x_dec, dec_ln2_g + i * HH, dec_ln2_b + i * HH, y_bf);
        gemm(BS, HH, HH, y_bf, HH, 0, 0, dwq2t + (size_t)i * HH2, HH, 0, 0,
             nullptr, 0, nullptr, qscale, nullptr, 0, q_bf, HH, 0, 0, 1, 1);
        gemm(BS, HH, HH, encout_bf, HH, 0, 0, dwk2t + (size_t)i * HH2, HH, 0, 0,
             nullptr, 0, nullptr, 1.f, nullptr, 0, k_bf, HH, 0, 0, 1, 1);
        gemm(BS, HH, HH, encout_bf, HH, 0, 0, dwv2t + (size_t)i * HH2, HH, 0, 0,
             nullptr, 0, nullptr, 1.f, nullptr, 0, v_bf, HH, 0, 0, 1, 1);
        transV(v_bf, vt_bf);
        gemm(TT, SS, DH, q_bf, HH, SH, DH, k_bf, HH, SH, DH,
             nullptr, 0, nullptr, 1.f, scores, SS, nullptr, 0,
             (long)NHEAD * SS2, SS2, BB * NHEAD, NHEAD);
        k_softmax<<<dim3(BB * NHEAD * TT), dim3(256), 0, stream>>>(scores, probs, inputs, TT, SS, 0, NHEAD);
        gemm(TT, DH, SS, probs, SS, (long)NHEAD * SS2, SS2, vt_bf, SS, HS, DS,
             nullptr, 0, nullptr, 1.f, nullptr, 0, o_bf, HH, SH, DH, BB * NHEAD, NHEAD);
        gemm(BS, HH, HH, o_bf, HH, 0, 0, dwo2t + (size_t)i * HH2, HH, 0, 0,
             nullptr, 0, x_dec, 1.f, x_dec, HH, nullptr, 0, 0, 0, 1, 1);
        // --- FFN ---
        k_ln<<<dim3(BS), dim3(256), 0, stream>>>(x_dec, dec_ln3_g + i * HH, dec_ln3_b + i * HH, y_bf);
        gemm(BS, FF, HH, y_bf, HH, 0, 0, dw1t + (size_t)i * HF, HH, 0, 0,
             dec_b1 + (size_t)i * FF, 1, nullptr, 1.f, nullptr, 0, h1_bf, FF, 0, 0, 1, 1);
        gemm(BS, HH, FF, h1_bf, FF, 0, 0, dw2t + (size_t)i * HF, FF, 0, 0,
             dec_b2 + (size_t)i * HH, 0, x_dec, 1.f, x_dec, HH, nullptr, 0, 0, 0, 1, 1);
    }
    k_ln<<<dim3(BS), dim3(256), 0, stream>>>(x_dec, dec_lnf_g, dec_lnf_b, y_bf);

    // ---- tied output projection: logits = dec_out @ emb^T (emb is already [N][K]) ----
    gemm(BS, VV, HH, y_bf, HH, 0, 0, emb_bf, HH, 0, 0,
         nullptr, 0, nullptr, 1.f, (float*)d_out, VV, nullptr, 0, 0, 0, 1, 1);
}
